// SignedSAGEConvolutionDeep_55525337202996
// MI455X (gfx1250) — matrix operand order repaired and checked by reading, not compiled, not run on hardware
//
#include <hip/hip_runtime.h>
#include <hip/hip_bf16.h>

#define N_NODES 6144
#define D_FEAT  64
#define FE_DIM  32
#define E_EDGES 196608
#define OUT_DIM 64
#define INDIM   384
#define WPR     192   // mask words per row = 6144/32

typedef __attribute__((ext_vector_type(16))) _Float16 v16h;
typedef __attribute__((ext_vector_type(8)))  float    v8f;
typedef __attribute__((ext_vector_type(2)))  __fp16   v2fp16;

union U16H { v16h v; unsigned int u[8]; };

static __device__ inline unsigned int pk2(float a, float b) {
    v2fp16 t = __builtin_amdgcn_cvt_pkrtz(a, b);
    return __builtin_bit_cast(unsigned int, t);
}

static __device__ inline v8f wmma16(v16h a, v16h b, v8f c) {
    return __builtin_amdgcn_wmma_f32_16x16x32_f16(false, a, false, b, (short)0, c,
                                                  false, false);
}

// A-fragment (16x32 f16), row-major source, ld halves per row.
// lane m<16: K = k0..k0+15 ; lane m>=16: K = k0+16..k0+31.
static __device__ inline v16h load_a16(const _Float16* base, int ld, int row0, int k0, int lane) {
    int m = lane & 15, hi = lane >> 4;
    const _Float16* p = base + (size_t)(row0 + m) * ld + k0 + hi * 16;
    U16H r;
    *reinterpret_cast<uint4*>(&r.u[0]) = *reinterpret_cast<const uint4*>(p);
    *reinterpret_cast<uint4*>(&r.u[4]) = *reinterpret_cast<const uint4*>(p + 8);
    return r.v;
}

// B-fragment (32x16 f16) where B[k][n] = src[nbase+n][k0+k] (row-major src).
// lane n'<16: K = k0..k0+15 ; lane>=16: K = k0+16..k0+31.
static __device__ inline v16h load_b16(const _Float16* base, int ld, int nbase, int k0, int lane) {
    int n = lane & 15, hi = lane >> 4;
    const _Float16* p = base + (size_t)(nbase + n) * ld + k0 + hi * 16;
    U16H r;
    *reinterpret_cast<uint4*>(&r.u[0]) = *reinterpret_cast<const uint4*>(p);
    *reinterpret_cast<uint4*>(&r.u[4]) = *reinterpret_cast<const uint4*>(p + 8);
    return r.v;
}

// A-fragment from f32 source, converted to f16 on the fly.
static __device__ inline v16h load_a32(const float* base, int ld, int row0, int k0, int lane) {
    int m = lane & 15, hi = lane >> 4;
    const float* p = base + (size_t)(row0 + m) * ld + k0 + hi * 16;
    U16H r;
#pragma unroll
    for (int q = 0; q < 4; q++) r.u[q] = pk2(p[2 * q], p[2 * q + 1]);
    const float* p2 = p + 8;
#pragma unroll
    for (int q = 0; q < 4; q++) r.u[4 + q] = pk2(p2[2 * q], p2[2 * q + 1]);
    return r.v;
}

static __device__ inline void store8(float* dst, v8f a, float s) {
    float4 u, v;
    u.x = a[0] * s; u.y = a[1] * s; u.z = a[2] * s; u.w = a[3] * s;
    v.x = a[4] * s; v.y = a[5] * s; v.z = a[6] * s; v.w = a[7] * s;
    *reinterpret_cast<float4*>(dst) = u;
    *reinterpret_cast<float4*>(dst + 4) = v;
}

// ---------------- utility kernels ----------------

__global__ void k_zero(unsigned int* p, long long n) {
    long long i = (long long)blockIdx.x * blockDim.x + threadIdx.x;
    if (i < n) p[i] = 0u;
}

struct WtArgs { const float* w[8]; _Float16* dst; };

// Transpose+convert 8 64x64 f32 weight matrices to f16 [out][in].
__global__ void k_wt(WtArgs a) {
    int idx = blockIdx.x * blockDim.x + threadIdx.x;
    if (idx >= 8 * 4096) return;
    int mat = idx >> 12, e = idx & 4095;
    int i = e >> 6, o = e & 63;
    a.dst[(size_t)mat * 4096 + (size_t)o * 64 + i] = (_Float16)a.w[mat][(size_t)i * 64 + o];
}

// Set mask bits for edges + full diagonal (self loops cover all diags).
__global__ void k_mask(const int* rows, const int* cols, unsigned int* mw) {
    int e = blockIdx.x * blockDim.x + threadIdx.x;
    if (e < E_EDGES) {
        int i = rows[e], j = cols[e];
        atomicOr(&mw[(size_t)i * WPR + (j >> 5)], 1u << (j & 31));
    } else {
        int t = e - E_EDGES;
        if (t < N_NODES) atomicOr(&mw[(size_t)t * WPR + (t >> 5)], 1u << (t & 31));
    }
}

__global__ void k_segsum(const int* rows, const int* cols, const float* x, float* sum) {
    long long idx = (long long)blockIdx.x * blockDim.x + threadIdx.x;
    if (idx >= (long long)E_EDGES * 64) return;
    int e = (int)(idx >> 6), f = (int)(idx & 63);
    atomicAdd(&sum[(size_t)rows[e] * 64 + f], x[(size_t)cols[e] * 64 + f]);
}

__global__ void k_segcnt(const int* rows, unsigned int* cnt) {
    int e = blockIdx.x * blockDim.x + threadIdx.x;
    if (e < E_EDGES) atomicAdd(&cnt[rows[e]], 1u);
}

// Max-pool nonnegative edge features per node (int-compare trick valid for >=0).
__global__ void k_poolmax(const int* idx, const float* feat, float* pool) {
    long long t = (long long)blockIdx.x * blockDim.x + threadIdx.x;
    if (t >= (long long)E_EDGES * FE_DIM) return;
    int e = (int)(t >> 5), f = (int)(t & 31);
    atomicMax(reinterpret_cast<int*>(&pool[(size_t)idx[e] * FE_DIM + f]),
              __float_as_int(feat[(size_t)e * FE_DIM + f]));
}

// ---------------- WMMA projection: out = x(f32)[N,64] @ W + b ----------------
// wt is f16 [64(out)][64(in)]. out_mode: 0 = f16 row-major, 1 = f16 transposed
// (Vt[64][N]), 2 = f32 row-major.
__global__ void k_proj(const float* x, const _Float16* wt, const float* bias,
                       void* out, int out_mode) {
    int lane = threadIdx.x & 31;
    int tile = blockIdx.x * (blockDim.x >> 5) + (threadIdx.x >> 5);
    if (tile >= N_NODES / 16) return;
    int row0 = tile * 16;
    int n = lane & 15, hi = lane >> 4;
    v16h a0 = load_a32(x, 64, row0, 0, lane);
    v16h a1 = load_a32(x, 64, row0, 32, lane);
#pragma unroll
    for (int oc = 0; oc < 4; oc++) {
        float bv = bias[oc * 16 + n];
        v8f c = {bv, bv, bv, bv, bv, bv, bv, bv};
        v16h b0 = load_b16(wt, 64, oc * 16, 0, lane);
        v16h b1 = load_b16(wt, 64, oc * 16, 32, lane);
        c = wmma16(a0, b0, c);
        c = wmma16(a1, b1, c);
        if (out_mode == 0) {
            _Float16* o = (_Float16*)out;
#pragma unroll
            for (int r = 0; r < 8; r++)
                o[(size_t)(row0 + r + 8 * hi) * 64 + oc * 16 + n] = (_Float16)c[r];
        } else if (out_mode == 1) {
            // Vt[oc*16+n][row0 + 8*hi + r] : 8 contiguous halves -> one b128 store
            unsigned int u0 = pk2(c[0], c[1]), u1 = pk2(c[2], c[3]);
            unsigned int u2 = pk2(c[4], c[5]), u3 = pk2(c[6], c[7]);
            uint4 st = {u0, u1, u2, u3};
            _Float16* o = (_Float16*)out;
            *reinterpret_cast<uint4*>(o + (size_t)(oc * 16 + n) * N_NODES + row0 + 8 * hi) = st;
        } else {
            float* o = (float*)out;
#pragma unroll
            for (int r = 0; r < 8; r++)
                o[(size_t)(row0 + r + 8 * hi) * 64 + oc * 16 + n] = c[r];
        }
    }
}

// ---------------- flash attention (transposed-S formulation) ----------------
// Sᵀ = K·Qᵀ so query index i lives in the lane dim: softmax stats are per-lane.
// Oᵀ = Vᵀ·Pᵀ accumulated in 4 C-tiles; Pᵀ B-operand built via 8 xor-16 shuffles.
__global__ void k_attn(const _Float16* __restrict__ Q, const _Float16* __restrict__ K,
                       const _Float16* __restrict__ Vt, const unsigned int* __restrict__ mw,
                       float* __restrict__ Oout) {
    int lane = threadIdx.x & 31;
    int tile = blockIdx.x * (blockDim.x >> 5) + (threadIdx.x >> 5);
    if (tile >= N_NODES / 16) return;
    int i0 = tile * 16;
    int n = lane & 15, hi = lane >> 4;

    v16h bq0 = load_b16(Q, 64, i0, 0, lane);   // Qᵀ  k=0..31
    v16h bq1 = load_b16(Q, 64, i0, 32, lane);  // Qᵀ  k=32..63

    v8f o0 = {0.f,0.f,0.f,0.f,0.f,0.f,0.f,0.f};
    v8f o1 = o0, o2 = o0, o3 = o0;
    float m_run = -1e30f, l_run = 0.f;
    const float scale = 0.125f;  // 1/sqrt(64)

    for (int j0 = 0; j0 < N_NODES; j0 += 32) {
        if (j0 + 32 < N_NODES)
            __builtin_prefetch(K + (size_t)(j0 + 32) * 64, 0, 3);
        // Sᵀ tiles: c0 -> j = j0+0..15, c1 -> j = j0+16..31 (cols i = i0+n)
        v8f c0 = {0.f,0.f,0.f,0.f,0.f,0.f,0.f,0.f};
        v8f c1 = c0;
        v16h ak;
        ak = load_a16(K, 64, j0,      0,  lane); c0 = wmma16(ak, bq0, c0);
        ak = load_a16(K, 64, j0,      32, lane); c0 = wmma16(ak, bq1, c0);
        ak = load_a16(K, 64, j0 + 16, 0,  lane); c1 = wmma16(ak, bq0, c1);
        ak = load_a16(K, 64, j0 + 16, 32, lane); c1 = wmma16(ak, bq1, c1);

        unsigned int w = mw[(size_t)(i0 + n) * WPR + (j0 >> 5)];
        float s[16];
#pragma unroll
        for (int r = 0; r < 8; r++) {
            int jb0 = r + 8 * hi;        // local j of c0[r]
            int jb1 = 16 + r + 8 * hi;   // local j of c1[r]
            s[r]     = c0[r] * scale + (((w >> jb0) & 1u) ? 0.f : 1.f);
            s[8 + r] = c1[r] * scale + (((w >> jb1) & 1u) ? 0.f : 1.f);
        }
        float tmax = s[0];
#pragma unroll
        for (int q = 1; q < 16; q++) tmax = fmaxf(tmax, s[q]);
        tmax = fmaxf(tmax, __shfl_xor(tmax, 16, 32));
        float m_new = fmaxf(m_run, tmax);
        float alpha = __expf(m_run - m_new);
        float p[16], lsum = 0.f;
#pragma unroll
        for (int q = 0; q < 16; q++) { p[q] = __expf(s[q] - m_new); lsum += p[q]; }
        lsum += __shfl_xor(lsum, 16, 32);
        l_run = l_run * alpha + lsum;
        m_run = m_new;
#pragma unroll
        for (int r = 0; r < 8; r++) {
            o0[r] *= alpha; o1[r] *= alpha; o2[r] *= alpha; o3[r] *= alpha;
        }
        // Build Pᵀ B-fragment: lane(hi,n) needs halves h -> j = 16*hi + h, i = n.
        U16H bp;
#pragma unroll
        for (int q = 0; q < 4; q++) {
            unsigned int own0 = pk2(p[2 * q], p[2 * q + 1]);
            unsigned int own1 = pk2(p[8 + 2 * q], p[9 + 2 * q]);
            unsigned int part0 = __shfl_xor(own0, 16, 32);
            unsigned int part1 = __shfl_xor(own1, 16, 32);
            bp.u[q]     = hi ? part1 : own0;
            bp.u[4 + q] = hi ? own1 : part0;
        }
        // Oᵀ += Vᵀ·Pᵀ  (A from Vt rows = feature chunks, K-dim = 32 keys)
        v16h av;
        av = load_a16(Vt, N_NODES, 0,  j0, lane); o0 = wmma16(av, bp.v, o0);
        av = load_a16(Vt, N_NODES, 16, j0, lane); o1 = wmma16(av, bp.v, o1);
        av = load_a16(Vt, N_NODES, 32, j0, lane); o2 = wmma16(av, bp.v, o2);
        av = load_a16(Vt, N_NODES, 48, j0, lane); o3 = wmma16(av, bp.v, o3);
    }
    float inv = 1.0f / l_run;  // additive 0/1 mask -> denominator always > 0
    float* orow = Oout + (size_t)(i0 + n) * 64;
    store8(orow + 0  + 8 * hi, o0, inv);
    store8(orow + 16 + 8 * hi, o1, inv);
    store8(orow + 32 + 8 * hi, o2, inv);
    store8(orow + 48 + 8 * hi, o3, inv);
}

// ---------------- feature assembly (one wave per node) ----------------
__global__ void k_feat(const float* x1, const float* x2,
                       const float* sp, const float* sn,
                       const unsigned int* cp, const unsigned int* cn,
                       const float* ap, const float* an,
                       const float* pool, float* feat) {
    int node = blockIdx.x * (blockDim.x >> 5) + (threadIdx.x >> 5);
    int lane = threadIdx.x & 31;
    if (node >= N_NODES) return;
    float* fr = feat + (size_t)node * INDIM;
    float icp = 1.f / (float)(cp[node] + 1u);
    float icn = 1.f / (float)(cn[node] + 1u);
#pragma unroll
    for (int q = 0; q < 2; q++) {
        int f = lane + 32 * q;
        size_t o = (size_t)node * 64 + f;
        fr[f]       = (sp[o] + x1[o]) * icp + ap[o];
        fr[64 + f]  = (sn[o] + x2[o]) * icn + an[o];
        fr[128 + f] = x1[o];
        fr[192 + f] = x2[o];
    }
#pragma unroll
    for (int g = 0; g < 4; g++) {
        float v = pool[(size_t)g * N_NODES * FE_DIM + (size_t)node * FE_DIM + lane];
        float ss = v * v;
#pragma unroll
        for (int off = 16; off; off >>= 1) ss += __shfl_xor(ss, off, 32);
        fr[256 + g * 32 + lane] = v / fmaxf(sqrtf(ss), 1e-12f);
    }
}

// ---------------- final 384->64 GEMM + row L2 norm (f32) ----------------
__global__ void k_final(const float* feat, const float* W, const float* B, float* out) {
    __shared__ float sf[INDIM];
    __shared__ float red[2];
    int node = blockIdx.x;
    int t = threadIdx.x;  // 64 threads
    for (int k = t; k < INDIM; k += 64) sf[k] = feat[(size_t)node * INDIM + k];
    __syncthreads();
    float acc = B[t];
    for (int k = 0; k < INDIM; k++) acc = fmaf(sf[k], W[(size_t)k * 64 + t], acc);
    float ss = acc * acc;
#pragma unroll
    for (int off = 16; off; off >>= 1) ss += __shfl_xor(ss, off, 32);
    if ((t & 31) == 0) red[t >> 5] = ss;
    __syncthreads();
    float nrm = sqrtf(red[0] + red[1]);
    out[(size_t)node * 64 + t] = acc / fmaxf(nrm, 1e-12f);
}

// ---------------- host orchestration ----------------

extern "C" void kernel_launch(void* const* d_in, const int* in_sizes, int n_in,
                              void* d_out, int out_size, void* d_ws, size_t ws_size,
                              hipStream_t stream) {
    (void)in_sizes; (void)n_in; (void)out_size; (void)ws_size;
    const float* x1  = (const float*)d_in[0];
    const float* x2  = (const float*)d_in[1];
    const int*   ep  = (const int*)d_in[2];   // [2][E]
    const int*   en  = (const int*)d_in[3];
    const float* efp = (const float*)d_in[4];
    const float* efn = (const float*)d_in[5];
    const float* W   = (const float*)d_in[6];
    const float* B   = (const float*)d_in[7];
    const float* Wm[8]; const float* bm[8];
    for (int i = 0; i < 8; i++) { Wm[i] = (const float*)d_in[8 + 2 * i];
                                  bm[i] = (const float*)d_in[9 + 2 * i]; }
    // Wm order: 0 q_p, 1 k_p, 2 v_p, 3 o_p, 4 q_n, 5 k_n, 6 v_n, 7 o_n

    char* base = (char*)d_ws; size_t off = 0;
    auto carve = [&](size_t b) -> void* {
        void* p = base + off; off = (off + b + 255) & ~(size_t)255; return p;
    };
    // --- zeroed region ---
    unsigned int* mw_p = (unsigned int*)carve((size_t)N_NODES * WPR * 4);
    unsigned int* mw_n = (unsigned int*)carve((size_t)N_NODES * WPR * 4);
    float* sum_p = (float*)carve((size_t)N_NODES * 64 * 4);
    float* sum_n = (float*)carve((size_t)N_NODES * 64 * 4);
    unsigned int* cnt_p = (unsigned int*)carve((size_t)N_NODES * 4);
    unsigned int* cnt_n = (unsigned int*)carve((size_t)N_NODES * 4);
    float* pool = (float*)carve((size_t)4 * N_NODES * FE_DIM * 4);
    size_t zero_words = off / 4;
    // --- fully-overwritten region ---
    _Float16* Qp  = (_Float16*)carve((size_t)N_NODES * 64 * 2);
    _Float16* Kp  = (_Float16*)carve((size_t)N_NODES * 64 * 2);
    _Float16* Vtp = (_Float16*)carve((size_t)N_NODES * 64 * 2);
    _Float16* Qn  = (_Float16*)carve((size_t)N_NODES * 64 * 2);
    _Float16* Kn  = (_Float16*)carve((size_t)N_NODES * 64 * 2);
    _Float16* Vtn = (_Float16*)carve((size_t)N_NODES * 64 * 2);
    float* Oraw_p = (float*)carve((size_t)N_NODES * 64 * 4);
    float* Oraw_n = (float*)carve((size_t)N_NODES * 64 * 4);
    float* attn_p = (float*)carve((size_t)N_NODES * 64 * 4);
    float* attn_n = (float*)carve((size_t)N_NODES * 64 * 4);
    float* feat   = (float*)carve((size_t)N_NODES * INDIM * 4);
    _Float16* wt16 = (_Float16*)carve((size_t)8 * 4096 * 2);

    // 1. zero accumulators / masks
    {
        long long nw = (long long)zero_words;
        int blk = (int)((nw + 255) / 256);
        k_zero<<<blk, 256, 0, stream>>>((unsigned int*)d_ws, nw);
    }
    // 2. weight transpose+f16
    {
        WtArgs wa; for (int i = 0; i < 8; i++) wa.w[i] = Wm[i]; wa.dst = wt16;
        k_wt<<<128, 256, 0, stream>>>(wa);
    }
    // 3. mask bits (edges + diagonal)
    {
        int nt = E_EDGES + N_NODES, blk = (nt + 255) / 256;
        k_mask<<<blk, 256, 0, stream>>>(ep, ep + E_EDGES, mw_p);
        k_mask<<<blk, 256, 0, stream>>>(en, en + E_EDGES, mw_n);
    }
    // 4. segment sums / counts
    {
        long long nt = (long long)E_EDGES * 64; int blk = (int)((nt + 255) / 256);
        k_segsum<<<blk, 256, 0, stream>>>(ep, ep + E_EDGES, x1, sum_p);
        k_segsum<<<blk, 256, 0, stream>>>(en, en + E_EDGES, x2, sum_n);
        int blk2 = (E_EDGES + 255) / 256;
        k_segcnt<<<blk2, 256, 0, stream>>>(ep, cnt_p);
        k_segcnt<<<blk2, 256, 0, stream>>>(en, cnt_n);
    }
    // 5. edge-feature max pools: po(row,pos) pi(col,pos) no(row,neg) ni(col,neg)
    {
        long long nt = (long long)E_EDGES * FE_DIM; int blk = (int)((nt + 255) / 256);
        k_poolmax<<<blk, 256, 0, stream>>>(ep,            efp, pool + 0 * (size_t)N_NODES * FE_DIM);
        k_poolmax<<<blk, 256, 0, stream>>>(ep + E_EDGES,  efp, pool + 1 * (size_t)N_NODES * FE_DIM);
        k_poolmax<<<blk, 256, 0, stream>>>(en,            efn, pool + 2 * (size_t)N_NODES * FE_DIM);
        k_poolmax<<<blk, 256, 0, stream>>>(en + E_EDGES,  efn, pool + 3 * (size_t)N_NODES * FE_DIM);
    }
    // 6. Q/K/V projections (WMMA, f16 out; V transposed)
    {
        int blk = (N_NODES / 16 + 3) / 4;  // 4 waves/block
        k_proj<<<blk, 128, 0, stream>>>(x1, wt16 + 0 * 4096, bm[0], Qp,  0);
        k_proj<<<blk, 128, 0, stream>>>(x1, wt16 + 1 * 4096, bm[1], Kp,  0);
        k_proj<<<blk, 128, 0, stream>>>(x1, wt16 + 2 * 4096, bm[2], Vtp, 1);
        k_proj<<<blk, 128, 0, stream>>>(x2, wt16 + 4 * 4096, bm[4], Qn,  0);
        k_proj<<<blk, 128, 0, stream>>>(x2, wt16 + 5 * 4096, bm[5], Kn,  0);
        k_proj<<<blk, 128, 0, stream>>>(x2, wt16 + 6 * 4096, bm[6], Vtn, 1);
    }
    // 7. flash attention (WMMA)
    {
        int blk = (N_NODES / 16 + 3) / 4;
        k_attn<<<blk, 128, 0, stream>>>(Qp, Kp, Vtp, mw_p, Oraw_p);
        k_attn<<<blk, 128, 0, stream>>>(Qn, Kn, Vtn, mw_n, Oraw_n);
    }
    // 8. output projections (WMMA, f32 out)
    {
        int blk = (N_NODES / 16 + 3) / 4;
        k_proj<<<blk, 128, 0, stream>>>(Oraw_p, wt16 + 3 * 4096, bm[3], attn_p, 2);
        k_proj<<<blk, 128, 0, stream>>>(Oraw_n, wt16 + 7 * 4096, bm[7], attn_n, 2);
    }
    // 9. feature assembly
    {
        int blk = (N_NODES + 3) / 4;
        k_feat<<<blk, 128, 0, stream>>>(x1, x2, sum_p, sum_n, cnt_p, cnt_n,
                                        attn_p, attn_n, pool, feat);
    }
    // 10. final GEMM + L2 norm
    k_final<<<N_NODES, 64, 0, stream>>>(feat, W, B, (float*)d_out);
}
